// Learned_HighDimBPDecoder_8744553415067
// MI455X (gfx1250) — matrix operand order repaired and checked by reading, not compiled, not run on hardware
//
#include <hip/hip_runtime.h>
#include <hip/hip_bf16.h>
#include <math.h>

typedef _Float16 h16;
typedef __attribute__((ext_vector_type(16))) _Float16 v16h;
typedef __attribute__((ext_vector_type(8)))  _Float16 v8h;
typedef __attribute__((ext_vector_type(8)))  float    v8f;

#define NUM_CHKS 512
#define NUM_VARS 1024
#define DC 6
#define DV 3
#define DD 16
#define TT 5
#define BATCH 256
#define EDGES (NUM_CHKS * DC)          // 3072

// ---- check MLP dims (in=97, hid=388, out=96) ----
#define C_IN   97
#define C_KP   128                      // padded K (4 k-steps of 32)
#define C_KT   4
#define C_HID  388
#define C_HP   416                      // padded hidden: 26 n-tiles, 13 k-steps
#define C_NT1  26
#define C_KT2  13
#define C_OUT  96
#define C_NT2  6
// ---- var MLP dims (in=49, hid=196, out=49) ----
#define V_IN   49
#define V_KP   64                       // 2 k-steps
#define V_KT   2
#define V_HID  196
#define V_HP   224                      // 14 n-tiles, 7 k-steps
#define V_NT1  14
#define V_KT2  7
#define V_OUT  49
#define V_OP   64                       // padded out
#define V_NT2  4

// ------------------------------------------------------------------
// CDNA5 async global->LDS copy (ASYNCcnt path), GV addressing form.
// LDS operand = low 32 bits of generic address (addr[31:0] == LDS offset).
// ------------------------------------------------------------------
__device__ __forceinline__ void async_ld_b128(const void* gsrc, void* ldst) {
  unsigned int  l = (unsigned int)(unsigned long long)(uintptr_t)ldst;
  unsigned long long g = (unsigned long long)(uintptr_t)gsrc;
  asm volatile("global_load_async_to_lds_b128 %0, %1, off"
               :: "v"(l), "v"(g) : "memory");
}
__device__ __forceinline__ void wait_async0() {
#if __has_builtin(__builtin_amdgcn_s_wait_asynccnt)
  __builtin_amdgcn_s_wait_asynccnt(0);
#else
  asm volatile("s_wait_asynccnt 0x0" ::: "memory");
#endif
}

// ------------------------------------------------------------------
// WMMA helpers (v_wmma_f32_16x16x32_f16, wave32)
// ------------------------------------------------------------------
__device__ __forceinline__ v8f wmma_f16(v16h a, v16h b, v8f c) {
  return __builtin_amdgcn_wmma_f32_16x16x32_f16(
      /*neg_a=*/false, a, /*neg_b=*/false, b,
      /*c_mod=*/(short)0, c, /*reuse_a=*/false, /*reuse_b=*/false);
}

// A-fragment (16x32 f16, MxK) from LDS, per documented layout:
// lane L: M = L&15, h = L>>4; halves 0..7 -> K = kBase + 8h + j,
//                              halves 8..15 -> K = kBase + 16 + 8h + j.
__device__ __forceinline__ v16h ldsA(const h16* s, int rowStride, int rowBase, int kBase) {
  int lane = threadIdx.x & 31;
  int m = lane & 15;
  int h = lane >> 4;
  const h16* p = s + (size_t)(rowBase + m) * rowStride + kBase + h * 8;
  v8h a0 = *(const v8h*)p;
  v8h a1 = *(const v8h*)(p + 16);
  return __builtin_shufflevector(a0, a1, 0,1,2,3,4,5,6,7,8,9,10,11,12,13,14,15);
}

// B-fragment from pre-packed weights: tile tIdx, lane L reads 16 contiguous halves.
__device__ __forceinline__ v16h ldB(const h16* packed, int tIdx) {
  int lane = threadIdx.x & 31;
  const h16* p = packed + ((size_t)tIdx * 32 + lane) * 16;
  v8h b0 = *(const v8h*)p;
  v8h b1 = *(const v8h*)(p + 8);
  return __builtin_shufflevector(b0, b1, 0,1,2,3,4,5,6,7,8,9,10,11,12,13,14,15);
}

// Branch-free tanh-GELU using gfx1250 hardware V_TANH_F32.
__device__ __forceinline__ float gelu_fast(float x) {
  float u = 0.7978845608028654f * x * (1.0f + 0.044715f * x * x);
#if __has_builtin(__builtin_amdgcn_tanhf)
  float t = __builtin_amdgcn_tanhf(u);
#else
  float e = __expf(-2.0f * u);                                  // v_exp_f32
  float t = 1.0f - 2.0f * e * __builtin_amdgcn_rcpf(1.0f + e);  // tanh(u)
#endif
  return 0.5f * x * (1.0f + t);
}

// ------------------------------------------------------------------
// Weight packing into WMMA B-fragment order.
// dst[(nt*KT + kt)*32 + lane][j] = W[kt*32 + (lane>>4)*8 + (j&7) + (j>>3)*16][nt*16 + (lane&15)]
// ------------------------------------------------------------------
__global__ void k_pack(const float* __restrict__ W, h16* __restrict__ dst,
                       int Kreal, int Nreal, int KT, int NT) {
  int tid = blockIdx.x * blockDim.x + threadIdx.x;
  int total = KT * NT * 32;
  if (tid >= total) return;
  int lane = tid & 31;
  int tIdx = tid >> 5;
  int kt = tIdx % KT;
  int nt = tIdx / KT;
  int n  = nt * 16 + (lane & 15);
  int kb = kt * 32 + (lane >> 4) * 8;
  h16* o = dst + (size_t)tid * 16;
#pragma unroll
  for (int j = 0; j < 16; ++j) {
    int k = kb + (j & 7) + (j >> 3) * 16;
    float v = (k < Kreal && n < Nreal) ? W[(size_t)k * Nreal + n] : 0.0f;
    o[j] = (h16)v;
  }
}

// Zero-padded bias copy (removes EXEC-divergent guards from GEMM epilogues).
__global__ void k_padbias(const float* __restrict__ b, float* __restrict__ dst,
                          int nreal, int npad) {
  int i = blockIdx.x * blockDim.x + threadIdx.x;
  if (i < npad) dst[i] = (i < nreal) ? b[i] : 0.0f;
}

__global__ void k_invperm(const int* __restrict__ perm, int* __restrict__ inv) {
  int i = blockIdx.x * blockDim.x + threadIdx.x;
  if (i < EDGES) inv[perm[i]] = i;
}

// Mc0[b,e,0] = prior[inv_perm[e]/DV], rest 0
__global__ void k_init(const int* __restrict__ inv, const float* __restrict__ prior,
                       h16* __restrict__ Mc) {
  int t = blockIdx.x * blockDim.x + threadIdx.x;
  if (t >= BATCH * EDGES) return;
  int e = t % EDGES;
  int voe = inv[e] / DV;
  h16* o = Mc + (size_t)t * DD;
  o[0] = (h16)prior[voe];
#pragma unroll
  for (int j = 1; j < DD; ++j) o[j] = (h16)0.0f;
}

// ------------------------------------------------------------------
// Check MLP: rows = b*512 + chk (131072 rows), 128 rows/block, 8 waves.
// yc = (gelu(x@W1+b1)@W2 + b2) * sgn, written f16 to Yc (B,E,16).
// b1 is zero-padded to C_HP floats.
// ------------------------------------------------------------------
__global__ void __launch_bounds__(256)
k_check(const h16* __restrict__ Mc, const int* __restrict__ synd,
        const h16* __restrict__ W1p, const float* __restrict__ b1,
        const h16* __restrict__ W2p, const float* __restrict__ b2,
        h16* __restrict__ Yc) {
  extern __shared__ h16 smem[];
  const int XST = C_KP + 8;             // 136 halves (272B, 16B aligned)
  const int HST = C_HP + 8;             // 424 halves
  h16* Xs = smem;                        // 128 x XST
  h16* Hs = smem + 128 * XST;            // 128 x HST

  int tid = threadIdx.x;
  int rowBlock = blockIdx.x * 128;       // global row base

  // ---- stage X tile via async global->LDS: [Mc row (96h) | sgn | zeros] ----
  for (int seg = tid; seg < 128 * 6; seg += 256) {   // 768 = 3 full passes
    int r = seg / 6, s6 = seg % 6;
    int grow = rowBlock + r;             // = b*512 + chk
    const h16* src = Mc + (size_t)grow * (DC * DD) + s6 * 16;
    h16* dst = Xs + (size_t)r * XST + s6 * 16;
    async_ld_b128(src,     dst);
    async_ld_b128(src + 8, dst + 8);
  }
  for (int r = tid; r < 128; r += 256) {
    int grow = rowBlock + r;
    float sg = 1.0f - 2.0f * (float)synd[grow];
    Xs[(size_t)r * XST + 96] = (h16)sg;
#pragma unroll
    for (int c = 97; c < C_KP; ++c) Xs[(size_t)r * XST + c] = (h16)0.0f;
  }
  wait_async0();
  __syncthreads();

  int wave = tid >> 5;                   // wave owns 16-row stripe
  int lane = tid & 31;
  int rbase = wave * 16;
  int h = lane >> 4;
  int ncol = lane & 15;

  // preload all 4 A-fragments (K = 128)
  v16h a0 = ldsA(Xs, XST, rbase, 0);
  v16h a1 = ldsA(Xs, XST, rbase, 32);
  v16h a2 = ldsA(Xs, XST, rbase, 64);
  v16h a3 = ldsA(Xs, XST, rbase, 96);

  // ---- layer 1: hidden (26 n-tiles x 4 WMMAs) -> gelu -> LDS ----
  for (int nt = 0; nt < C_NT1; ++nt) {
    v8f acc = {};
    acc = wmma_f16(a0, ldB(W1p, nt * C_KT + 0), acc);
    acc = wmma_f16(a1, ldB(W1p, nt * C_KT + 1), acc);
    acc = wmma_f16(a2, ldB(W1p, nt * C_KT + 2), acc);
    acc = wmma_f16(a3, ldB(W1p, nt * C_KT + 3), acc);
    int col = nt * 16 + ncol;
    float bb = b1[col];                  // zero-padded, no guard
#pragma unroll
    for (int r = 0; r < 8; ++r) {
      float v = gelu_fast(acc[r] + bb);  // padded cols: gelu(0)=0
      Hs[(size_t)(rbase + r + 8 * h) * HST + col] = (h16)v;
    }
  }
  __syncthreads();

  // per-row sign, hoisted out of the n-tile loop
  float sgn8[8];
#pragma unroll
  for (int r = 0; r < 8; ++r)
    sgn8[r] = 1.0f - 2.0f * (float)synd[rowBlock + rbase + r + 8 * h];

  // ---- layer 2: out (6 n-tiles x 13 WMMAs), *sgn, write Yc ----
  for (int nt = 0; nt < C_NT2; ++nt) {
    v8f acc = {};
#pragma unroll
    for (int ks = 0; ks < C_KT2; ++ks) {
      v16h a = ldsA(Hs, HST, rbase, ks * 32);
      acc = wmma_f16(a, ldB(W2p, nt * C_KT2 + ks), acc);
    }
    float bb = b2[nt * 16 + ncol];       // n-tile == edge slot j
#pragma unroll
    for (int r = 0; r < 8; ++r) {
      int row = rowBlock + rbase + r + 8 * h;           // b*512 + chk
      float v = (acc[r] + bb) * sgn8[r];
      Yc[(size_t)row * (DC * DD) + nt * 16 + ncol] = (h16)v;  // (b,E,16)
    }
  }
}

// ------------------------------------------------------------------
// Var MLP: rows = b*1024 + v (262144 rows). Gathers Yc via perm (async->LDS),
// writes cols 0..47 back to Mc via perm (fused inverse-permute),
// col 48 (+bias) -> llr output slice. b1 padded to V_HP, b2 padded to V_OP.
// ------------------------------------------------------------------
__global__ void __launch_bounds__(256)
k_var(const h16* __restrict__ Yc, const float* __restrict__ prior,
      const int* __restrict__ perm,
      const h16* __restrict__ W1p, const float* __restrict__ b1,
      const h16* __restrict__ W2p, const float* __restrict__ b2,
      h16* __restrict__ Mc, float* __restrict__ llr_out) {
  extern __shared__ h16 smem[];
  const int XST = V_KP + 8;             // 72 halves (144B)
  const int HST = V_HP + 8;             // 232 halves
  h16* Xs = smem;
  h16* Hs = smem + 128 * XST;

  int tid = threadIdx.x;
  int rowBlock = blockIdx.x * 128;

  // ---- stage X tile: async-gather 3 edges of 16 + prior + zeros ----
  for (int seg = tid; seg < 128 * 3; seg += 256) {
    int r = seg / 3, j = seg % 3;
    int grow = rowBlock + r;
    int b = grow >> 10;
    int v = grow & 1023;
    int pe = perm[v * DV + j];
    const h16* src = Yc + ((size_t)b * EDGES + pe) * DD;
    h16* dst = Xs + (size_t)r * XST + j * 16;
    async_ld_b128(src,     dst);
    async_ld_b128(src + 8, dst + 8);
  }
  for (int r = tid; r < 128; r += 256) {
    int grow = rowBlock + r;
    int v = grow & 1023;
    Xs[(size_t)r * XST + 48] = (h16)prior[v];
#pragma unroll
    for (int c = 49; c < V_KP; ++c) Xs[(size_t)r * XST + c] = (h16)0.0f;
  }
  wait_async0();
  __syncthreads();

  int wave = tid >> 5;
  int lane = tid & 31;
  int rbase = wave * 16;
  int h = lane >> 4;
  int ncol = lane & 15;

  v16h a0 = ldsA(Xs, XST, rbase, 0);
  v16h a1 = ldsA(Xs, XST, rbase, 32);

  // ---- layer 1: 14 n-tiles x 2 WMMAs ----
  for (int nt = 0; nt < V_NT1; ++nt) {
    v8f acc = {};
    acc = wmma_f16(a0, ldB(W1p, nt * V_KT + 0), acc);
    acc = wmma_f16(a1, ldB(W1p, nt * V_KT + 1), acc);
    int col = nt * 16 + ncol;
    float bb = b1[col];                  // zero-padded, no guard
#pragma unroll
    for (int r = 0; r < 8; ++r) {
      float v = gelu_fast(acc[r] + bb);
      Hs[(size_t)(rbase + r + 8 * h) * HST + col] = (h16)v;
    }
  }
  __syncthreads();

  // ---- layer 2: 4 n-tiles x 7 WMMAs ----
  for (int nt = 0; nt < V_NT2; ++nt) {
    v8f acc = {};
#pragma unroll
    for (int ks = 0; ks < V_KT2; ++ks) {
      v16h a = ldsA(Hs, HST, rbase, ks * 32);
      acc = wmma_f16(a, ldB(W2p, nt * V_KT2 + ks), acc);
    }
    float bb = b2[nt * 16 + ncol];       // zero-padded to V_OP
    if (nt < 3) {
#pragma unroll
      for (int r = 0; r < 8; ++r) {
        int grow = rowBlock + rbase + r + 8 * h;
        int b = grow >> 10, v = grow & 1023;
        int pe = perm[v * DV + nt];                   // Mc_new[b, perm[e']] = A[b, e']
        Mc[((size_t)b * EDGES + pe) * DD + ncol] = (h16)(acc[r] + bb);
      }
    } else {
      if (ncol == 0) {                                // col 48 = llr; 49..63 dropped
#pragma unroll
        for (int r = 0; r < 8; ++r) {
          int grow = rowBlock + rbase + r + 8 * h;
          llr_out[grow] = acc[r] + bb;
        }
      }
    }
  }
}

// ------------------------------------------------------------------
extern "C" void kernel_launch(void* const* d_in, const int* in_sizes, int n_in,
                              void* d_out, int out_size, void* d_ws, size_t ws_size,
                              hipStream_t stream) {
  (void)in_sizes; (void)n_in; (void)out_size; (void)ws_size;
  const int*   synd  = (const int*)  d_in[0];
  // d_in[1] = num_iters (== TT, hardcoded)
  const float* prior = (const float*)d_in[2];
  const int*   perm  = (const int*)  d_in[3];
  const float* cW1 = (const float*)d_in[4];
  const float* cb1 = (const float*)d_in[5];
  const float* cW2 = (const float*)d_in[6];
  const float* cb2 = (const float*)d_in[7];
  const float* vW1 = (const float*)d_in[8];
  const float* vb1 = (const float*)d_in[9];
  const float* vW2 = (const float*)d_in[10];
  const float* vb2 = (const float*)d_in[11];
  float* out = (float*)d_out;

  // ---- workspace carve ----
  char* ws = (char*)d_ws;
  size_t off = 0;
  auto carve = [&](size_t bytes) -> void* {
    void* p = ws + off;
    off += (bytes + 255) & ~(size_t)255;
    return p;
  };
  int* inv  = (int*)carve((size_t)EDGES * sizeof(int));
  h16* Mc   = (h16*)carve((size_t)BATCH * EDGES * DD * sizeof(h16));   // ~25 MB
  h16* Yc   = (h16*)carve((size_t)BATCH * EDGES * DD * sizeof(h16));   // ~25 MB
  h16* cW1p = (h16*)carve((size_t)C_NT1 * C_KT  * 512 * sizeof(h16));
  h16* cW2p = (h16*)carve((size_t)C_NT2 * C_KT2 * 512 * sizeof(h16));
  h16* vW1p = (h16*)carve((size_t)V_NT1 * V_KT  * 512 * sizeof(h16));
  h16* vW2p = (h16*)carve((size_t)V_NT2 * V_KT2 * 512 * sizeof(h16));
  float* cb1p = (float*)carve((size_t)C_HP  * sizeof(float));
  float* cb2p = (float*)carve((size_t)C_OUT * sizeof(float));
  float* vb1p = (float*)carve((size_t)V_HP  * sizeof(float));
  float* vb2p = (float*)carve((size_t)V_OP  * sizeof(float));

  // ---- pack weights into WMMA B-fragment order, pad biases ----
  auto pack = [&](const float* W, h16* dst, int Kr, int Nr, int KT, int NT) {
    int total = KT * NT * 32;
    k_pack<<<(total + 255) / 256, 256, 0, stream>>>(W, dst, Kr, Nr, KT, NT);
  };
  pack(cW1, cW1p, C_IN,  C_HID, C_KT,  C_NT1);
  pack(cW2, cW2p, C_HID, C_OUT, C_KT2, C_NT2);
  pack(vW1, vW1p, V_IN,  V_HID, V_KT,  V_NT1);
  pack(vW2, vW2p, V_HID, V_OUT, V_KT2, V_NT2);
  k_padbias<<<(C_HP  + 255) / 256, 256, 0, stream>>>(cb1, cb1p, C_HID, C_HP);
  k_padbias<<<(C_OUT + 255) / 256, 256, 0, stream>>>(cb2, cb2p, C_OUT, C_OUT);
  k_padbias<<<(V_HP  + 255) / 256, 256, 0, stream>>>(vb1, vb1p, V_HID, V_HP);
  k_padbias<<<(V_OP  + 255) / 256, 256, 0, stream>>>(vb2, vb2p, V_OUT, V_OP);

  k_invperm<<<(EDGES + 255) / 256, 256, 0, stream>>>(perm, inv);
  k_init<<<(BATCH * EDGES + 255) / 256, 256, 0, stream>>>(inv, prior, Mc);

  size_t shmC = ((size_t)128 * (C_KP + 8) + (size_t)128 * (C_HP + 8)) * sizeof(h16); // ~140 KB
  size_t shmV = ((size_t)128 * (V_KP + 8) + (size_t)128 * (V_HP + 8)) * sizeof(h16); // ~76 KB

  for (int t = 0; t < TT; ++t) {
    k_check<<<(BATCH * NUM_CHKS) / 128, 256, shmC, stream>>>(
        Mc, synd, cW1p, cb1p, cW2p, cb2p, Yc);
    k_var<<<(BATCH * NUM_VARS) / 128, 256, shmV, stream>>>(
        Yc, prior, perm, vW1p, vb1p, vW2p, vb2p,
        Mc, out + (size_t)t * BATCH * NUM_VARS);
  }
}
